// MinimalNetwork_37529424232914
// MI455X (gfx1250) — compile-verified, matrix operand order checked by reading, not run on hardware
//
#include <hip/hip_runtime.h>
#include <hip/hip_bf16.h>

typedef __attribute__((ext_vector_type(16))) _Float16 v16h;
typedef __attribute__((ext_vector_type(8)))  float    v8f;

#define N_EDGES   200000
#define N_NODES   12500
#define FEAT_DIM  72
#define SH_DIM    25
#define R_DIM     1216
#define CC_TOTAL  1225
#define TILE      128     // edges per block
#define KPAD      112     // H=100 padded to 7 N-tiles of 16
#define SPACING   0.2777777778f
#define INV_SPACING 3.6f

__constant__ int   c_R_OFF[10]   = {0,64,128,192,256,448,640,704,896,1216};
__constant__ int   c_FEAT_OFF[4] = {0,8,32,72};
__constant__ int   c_NL[9]       = {1,1,1,1,3,3,1,3,5};
__constant__ int   c_CCOFF[9][5] = {
    {0,0,0,0,0},{1,0,0,0,0},{10,0,0,0,0},{35,0,0,0,0},
    {44,53,80,0,0},{125,170,245,0,0},{350,0,0,0,0},
    {375,420,495,0,0},{600,625,700,825,1000}};
__constant__ float c_NORM[3] = {0.72360125f, 0.82048483f, 0.93416586f};

// ---- WMMA operand loaders -------------------------------------------------
// 16-bit A 16x32 layout: lanes 0-15 M=lane K={0..7,16..23}; lanes 16-31 M=lane-16 K={8..15,24..31}
__device__ inline v16h load_a_lds(const _Float16* s, int stride, int mtile,
                                  int k0, int kmax, int lane) {
    int m  = mtile * 16 + (lane & 15);
    int kb = k0 + ((lane & 16) ? 8 : 0);
    v16h a;
#pragma unroll
    for (int t = 0; t < 8; ++t) {
        int k1 = kb + t, k2 = kb + 16 + t;
        a[t]     = (k1 < kmax) ? s[m * stride + k1] : (_Float16)0.0f;
        a[t + 8] = (k2 < kmax) ? s[m * stride + k2] : (_Float16)0.0f;
    }
    return a;
}

// B (32x16, K-major in memory: row-major W[K][N]), f32 -> f16 on the fly
__device__ inline v16h load_b_dense(const float* __restrict__ W, int ldw,
                                    int k0, int n0, int kmax, int nmax, int lane) {
    int n  = n0 + (lane & 15);
    int kb = k0 + ((lane & 16) ? 8 : 0);
    bool nok = (n < nmax);
    v16h b;
#pragma unroll
    for (int t = 0; t < 8; ++t) {
        int k1 = kb + t, k2 = kb + 16 + t;
        b[t]     = (nok && k1 < kmax) ? (_Float16)W[k1 * ldw + n] : (_Float16)0.0f;
        b[t + 8] = (nok && k2 < kmax) ? (_Float16)W[k2 * ldw + n] : (_Float16)0.0f;
    }
    return b;
}

// W3 B-tile with permuted column mapping: chunk column nIdx -> (q = cq*4 + nIdx/8, u = nIdx%8),
// q = k*8+v, W3 col = R_OFF[p] + (u*8+v)*nl + k
__device__ inline v16h load_b_w3(const float* __restrict__ W3, int p, int cq,
                                 int nl, int nt, int k0, int lane) {
    int nIdx = nt * 16 + (lane & 15);
    int qg   = cq * 4 + (nIdx >> 3);
    int u    = nIdx & 7;
    int kk   = qg >> 3;
    int v    = qg & 7;
    int col  = c_R_OFF[p] + (u * 8 + v) * nl + kk;
    int kb   = k0 + ((lane & 16) ? 8 : 0);
    v16h b;
#pragma unroll
    for (int t = 0; t < 8; ++t) {
        int k1 = kb + t, k2 = kb + 16 + t;
        b[t]     = (k1 < 100) ? (_Float16)W3[k1 * R_DIM + col] : (_Float16)0.0f;
        b[t + 8] = (k2 < 100) ? (_Float16)W3[k2 * R_DIM + col] : (_Float16)0.0f;
    }
    return b;
}

// swish via hardware reciprocal: avoids IEEE div expansion (v_div_scale/fixup)
__device__ inline float swishf(float x) {
    float e = __expf(-x);
    return 1.679177f * x * __builtin_amdgcn_rcpf(1.0f + e);
}

// one MLP layer: [128, K] @ W[K,100] -> swish -> sOut [128][112] f16
__device__ inline void run_layer(const _Float16* sIn, int strideIn, int kmaxA,
                                 const float* __restrict__ W, int kmaxB,
                                 int kTiles, _Float16* sOut, float scale,
                                 int wave, int lane) {
    for (int nt = 0; nt < 7; ++nt) {
        v8f acc = {0.f,0.f,0.f,0.f,0.f,0.f,0.f,0.f};
        for (int kt = 0; kt < kTiles; ++kt) {
            v16h a = load_a_lds(sIn, strideIn, wave, kt * 32, kmaxA, lane);
            v16h b = load_b_dense(W, 100, kt * 32, nt * 16, kmaxB, 100, lane);
            acc = __builtin_amdgcn_wmma_f32_16x16x32_f16(
                false, a, false, b, (short)0, acc, false, false);
        }
        int n    = nt * 16 + (lane & 15);
        int mrow = wave * 16 + ((lane & 16) ? 8 : 0);
#pragma unroll
        for (int r = 0; r < 8; ++r)
            sOut[(mrow + r) * KPAD + n] = (_Float16)swishf(acc[r] * scale);
    }
}

__global__ void __launch_bounds__(256)
zero_out_kernel(float* __restrict__ out, int n) {
    int i = blockIdx.x * blockDim.x + threadIdx.x;
    if (i < n) out[i] = 0.0f;
}

__global__ void __launch_bounds__(256)
fused_tp_kernel(const int* __restrict__ ei, const float* __restrict__ feat,
                const float* __restrict__ radii, const float* __restrict__ rsh,
                const float* __restrict__ cc,
                const float* __restrict__ W0, const float* __restrict__ W1,
                const float* __restrict__ W2, const float* __restrict__ W3,
                float* __restrict__ out) {
    __shared__ __align__(16) _Float16 sH0[TILE * KPAD];
    __shared__ __align__(16) _Float16 sH1[TILE * KPAD];
    __shared__ float sCC[CC_TOTAL];

    const int tid  = threadIdx.x;
    const int lane = tid & 31;
    const int wave = tid >> 5;
    const int e0   = blockIdx.x * TILE;

    for (int i = tid; i < CC_TOTAL; i += 256) sCC[i] = cc[i];

    // ---- gaussian basis into sH0 as [128][32] (cols 10..31 zero) ----------
    if (tid < TILE) {
        int e  = e0 + tid;
        int ec = (e < N_EDGES) ? e : (N_EDGES - 1);
        float r = radii[ec];
        _Float16* row = sH0 + tid * 32;
#pragma unroll
        for (int c = 0; c < 32; ++c) {
            if (c < 10) {
                float d = (r - (0.7f + (float)c * SPACING)) * INV_SPACING;
                row[c] = (_Float16)__expf(-d * d);
            } else {
                row[c] = (_Float16)0.0f;
            }
        }
    }
    __syncthreads();

    // ---- MLP: 3 swish layers (WMMA f16, f32 accumulate) -------------------
    run_layer(sH0, 32,   32,  W0, 10,  1, sH1, 0.31622776601f, wave, lane);
    __syncthreads();
    run_layer(sH1, KPAD, KPAD, W1, 100, 4, sH0, 0.1f, wave, lane);
    __syncthreads();
    run_layer(sH0, KPAD, KPAD, W2, 100, 4, sH1, 0.1f, wave, lane); // h3 -> sH1
    __syncthreads();

    float* sR = (float*)sH0;   // h2 is dead; reuse as R chunk [128][32] f32

    // ---- per-edge consumer state ------------------------------------------
    int e   = e0 + (tid & (TILE - 1));
    bool ev = (tid < TILE) && (e < N_EDGES);
    int ec  = (e < N_EDGES) ? e : (N_EDGES - 1);
    int src = 0, dst = 0;
    float y[SH_DIM];
    if (tid < TILE) {
        src = ei[ec];
        dst = ei[N_EDGES + ec];
        for (int f = 0; f < SH_DIM; ++f) y[f] = rsh[ec * SH_DIM + f];
    }
    float accv[40], fblk[40], T[25];

    // ---- W3 chunks + CG tensor product ------------------------------------
    for (int p = 0; p < 9; ++p) {
        const int i_  = p / 3, j_ = p % 3;
        const int lo2 = 2 * i_ + 1, li2 = 2 * j_ + 1;
        const int nl  = c_NL[p];
        const int lf0 = (i_ > j_) ? (i_ - j_) : (j_ - i_);

        if (tid < TILE) {
            if (j_ == 0)
                for (int t = 0; t < 40; ++t) accv[t] = 0.0f;
            int fb = c_FEAT_OFF[j_];
            for (int t = 0; t < 8 * li2; ++t)
                fblk[t] = feat[src * FEAT_DIM + fb + t];
        }
        int kprev = -1;

        for (int cq = 0; cq < 2 * nl; ++cq) {
            __syncthreads();   // sR free (previous consume done)
            // produce R chunk [128][32]: wave = M-tile, 2 N-tiles, K=128
            for (int nt = 0; nt < 2; ++nt) {
                v8f racc = {0.f,0.f,0.f,0.f,0.f,0.f,0.f,0.f};
                for (int kt = 0; kt < 4; ++kt) {
                    // prefetch next K-tile rows of W3 into cache (global_prefetch_b8)
                    if (kt < 3) {
                        int pfk = (kt + 1) * 32 + (lane & 31);
                        if (pfk < 100)
                            __builtin_prefetch(&W3[pfk * R_DIM + c_R_OFF[p]], 0, 3);
                    }
                    v16h a = load_a_lds(sH1, KPAD, wave, kt * 32, KPAD, lane);
                    v16h b = load_b_w3(W3, p, cq, nl, nt, kt * 32, lane);
                    racc = __builtin_amdgcn_wmma_f32_16x16x32_f16(
                        false, a, false, b, (short)0, racc, false, false);
                }
                int ncol = nt * 16 + (lane & 15);
                int mrow = wave * 16 + ((lane & 16) ? 8 : 0);
#pragma unroll
                for (int r = 0; r < 8; ++r)
                    sR[(mrow + r) * 32 + ncol] = racc[r] * 0.1f;
            }
            __syncthreads();   // sR visible

            // consume: one thread per edge
            if (tid < TILE) {
                for (int qi = 0; qi < 4; ++qi) {
                    int qg = cq * 4 + qi;
                    int kk = qg >> 3, v = qg & 7;
                    int lf = lf0 + kk;
                    int nf = 2 * lf + 1;
                    if (kk != kprev) {      // T[o][i2] = sum_f Y[f] * C[o,i2,f]
                        int co = c_CCOFF[p][kk];
                        for (int o = 0; o < lo2; ++o)
                            for (int i2 = 0; i2 < li2; ++i2) {
                                float s = 0.0f;
                                for (int f = 0; f < nf; ++f)
                                    s += y[lf * lf + f] *
                                         sCC[co + (o * li2 + i2) * nf + f];
                                T[o * li2 + i2] = s;
                            }
                        kprev = kk;
                    }
                    float g[5];             // g[o] = sum_i2 F[v,i2] * T[o,i2]
                    for (int o = 0; o < lo2; ++o) {
                        float s = 0.0f;
                        for (int i2 = 0; i2 < li2; ++i2)
                            s += fblk[v * li2 + i2] * T[o * li2 + i2];
                        g[o] = s;
                    }
#pragma unroll
                    for (int u = 0; u < 8; ++u) {
                        float rv = sR[(tid) * 32 + qi * 8 + u];
                        for (int o = 0; o < lo2; ++o)
                            accv[u * lo2 + o] += rv * g[o];
                    }
                }
            }
        }

        // flush i-block (after j==2) with norm, scatter to destination node
        if (tid < TILE && j_ == 2 && ev) {
            float nrm = c_NORM[i_];
            int base  = dst * FEAT_DIM + c_FEAT_OFF[i_];
            for (int u = 0; u < 8; ++u)
                for (int o = 0; o < lo2; ++o)
                    unsafeAtomicAdd(&out[base + u * lo2 + o],
                                    nrm * accv[u * lo2 + o]);
        }
    }
}

extern "C" void kernel_launch(void* const* d_in, const int* in_sizes, int n_in,
                              void* d_out, int out_size, void* d_ws, size_t ws_size,
                              hipStream_t stream) {
    const int*   ei    = (const int*)d_in[0];
    const float* feat  = (const float*)d_in[1];
    const float* radii = (const float*)d_in[2];
    const float* rsh   = (const float*)d_in[3];
    const float* cc    = (const float*)d_in[4];
    const float* W0    = (const float*)d_in[5];
    const float* W1    = (const float*)d_in[6];
    const float* W2    = (const float*)d_in[7];
    const float* W3    = (const float*)d_in[8];
    float* out = (float*)d_out;

    zero_out_kernel<<<(out_size + 255) / 256, 256, 0, stream>>>(out, out_size);

    int nblocks = (N_EDGES + TILE - 1) / TILE;   // 1563
    fused_tp_kernel<<<nblocks, 256, 0, stream>>>(ei, feat, radii, rsh, cc,
                                                 W0, W1, W2, W3, out);
}
